// Nequip_12738873000711
// MI455X (gfx1250) — matrix operand order repaired and checked by reading, not compiled, not run on hardware
//
// NequIP-style GNN forward + hand-derived backward (forces) for MI455X (gfx1250).
// Dense matmuls: wave32 WMMA f16 GEMM (v_wmma_f32_16x16x32_f16), double-buffered
// LDS staging; weight tiles staged by the Tensor Data Mover (tensor_load_to_lds,
// TENSORcnt) when the builtin is available, else vector global->LDS copies.
// Edge scatter via f32 atomics; radial-MLP activations recomputed in backward.
//
// Input flattening assumption (jax pytree: top-level dict insertion order,
// nested params dict keys sorted alphabetically):
//  0 positions | 1 atom_energies |
//  layer0: 2 w1_s, 3 w2_s, 4 w2_v, 5-8 w_rad[0..3], 9 wsk_s
//  layer1: 10 w1_s, 11 w1_v, 12 w2_s, 13 w2_v, 14-17 w_rad, 18 wsk_s, 19 wsk_v
//  layer2: 20 w1_s, 21 w1_v, 22 w2_s, 23 w2_v, 24-27 w_rad, 28 wsk_s, 29 wsk_v
//  30 w_out | 31 species | 32 senders | 33 receivers | 34 graph_idx
// d_out: [4 graph energies][N*3 forces] (float32).

#include <hip/hip_runtime.h>
#include <math.h>

typedef __attribute__((ext_vector_type(16))) _Float16 v16h;
typedef __attribute__((ext_vector_type(8)))  float    v8f;
typedef __attribute__((ext_vector_type(2)))  _Float16 f16x2;
typedef __attribute__((ext_vector_type(8)))  _Float16 f16x8;

#if defined(__gfx1250__) && __has_builtin(__builtin_amdgcn_tensor_load_to_lds) && \
    __has_builtin(__builtin_amdgcn_s_wait_tensorcnt)
#define HAVE_TDM 1
#else
#define HAVE_TDM 0
#endif

#define F_ACC   1
#define F_SILU  2

constexpr int   Nn = 3000;
constexpr int   Ee = 48000;
constexpr int   Hh = 128;
constexpr int   NP = 3008;                 // Nn rounded up to 16
constexpr float RMAXf      = 5.0f;
constexpr float PIf        = 3.14159265358979323846f;
constexpr float SQRT3f     = 1.7320508075688772f;
constexpr float INV_SQRT3f = 0.5773502691896258f;
constexpr float INVSQH     = 0.08838834764831845f; // 1/sqrt(128)
constexpr float INVSQ8     = 0.35355339059327373f; // 1/sqrt(8)
constexpr float AINV       = 0.25f;                // 1/sqrt(avg_n=16)
constexpr float W2SC       = 0.0625f;              // 1/sqrt(256)

__device__ __forceinline__ float siluf(float x) {
    float s = 1.0f / (1.0f + __expf(-x));
    return x * s;
}
__device__ __forceinline__ float dsiluf(float x) {
    float s = 1.0f / (1.0f + __expf(-x));
    return s * (1.0f + x * (1.0f - s));
}
__device__ __forceinline__ void atomAddF(float* p, float v) { atomicAdd(p, v); }

// ---------------------------------------------------------------------------
// Weight pack: f32 [R][C] -> padded f16.
//  trans=1: out[ro][co] = in[co][ro]  (column-major of in; dims Ro>=C, Co>=R)
//  trans=0: out[ro][co] = in[ro][co]  (row-major copy;      dims Ro>=R, Co>=C)
// ---------------------------------------------------------------------------
__global__ __launch_bounds__(256) void k_pack(
    const float* __restrict__ in, _Float16* __restrict__ out,
    int R, int C, int Ro, int Co, int trans)
{
    int i = blockIdx.x * 256 + threadIdx.x;
    if (i >= Ro * Co) return;
    int ro = i / Co, co = i % Co;
    float v = 0.0f;
    if (trans) { if (co < R && ro < C) v = in[(size_t)co * C + ro]; }
    else       { if (ro < R && co < C) v = in[(size_t)ro * C + co]; }
    out[(size_t)ro * Co + co] = (_Float16)v;
}

// ---------------------------------------------------------------------------
// Tensor Data Mover descriptor issue: load a [128 rows x 32 f16] tile of the
// packed weight (row stride K f16) into LDS with pitch 40 f16 (pad 16B per
// 64B row -> pad_interval code 3, pad_amount code 3). D# layout per CDNA5 ISA
// 8.3/8.4 (group0: count/lds/global/type; group1: dims/tile/pad/strides).
// ---------------------------------------------------------------------------
#if HAVE_TDM
typedef unsigned int tdm_u4 __attribute__((ext_vector_type(4)));
typedef int          tdm_i8 __attribute__((ext_vector_type(8)));
typedef int          tdm_i4 __attribute__((ext_vector_type(4)));

__device__ __forceinline__ void tdm_load_b(const _Float16* gptr, unsigned lds, int K)
{
    unsigned long long ga = (unsigned long long)(uintptr_t)gptr;
    tdm_u4 g0;
    g0[0] = 1u;                                          // count=1 (valid D#)
    g0[1] = lds;                                         // lds_addr (bytes)
    g0[2] = (unsigned)(ga & 0xffffffffu);                // global_addr[31:0]
    g0[3] = (unsigned)((ga >> 32) & 0x01ffffffu)         // global_addr[56:32]
          | (2u << 30);                                  // type=2 ("image")
    tdm_i8 g1;
    g1[0] = (1 << 16)      // data_size = 2 bytes
          | (1 << 20)      // pad_enable
          | (3 << 22)      // pad_interval: 16 DWORDs (64B) per row
          | (3 << 25);     // pad_amount: 4 DWORDs (16B)
    g1[1] = (K & 0xffff) << 16;          // tensor_dim0[15:0] = K
    g1[2] = (128 << 16);                 // tensor_dim0[31:16]=0 | tensor_dim1[15:0]=128
    g1[3] = (32 << 16);                  // tensor_dim1[31:16]=0 | tile_dim0=32
    g1[4] = 128;                         // tile_dim1=128, tile_dim2=0
    g1[5] = K;                           // tensor_dim0_stride[31:0] = K
    g1[6] = 0;
    g1[7] = 0;
    tdm_i4 z4 = {0, 0, 0, 0};
#if defined(__clang_major__) && (__clang_major__ >= 23)
    tdm_i8 z8 = {0, 0, 0, 0, 0, 0, 0, 0};
    __builtin_amdgcn_tensor_load_to_lds(g0, g1, z4, z4, z8, 0);
#else
    __builtin_amdgcn_tensor_load_to_lds(g0, g1, z4, z4, 0);
#endif
}
#endif

// ---------------------------------------------------------------------------
// WMMA GEMM: C[M,Nl] (=|+=) scale * A[M,K] @ B, optional fused SiLU.
// Requirements: M % 16 == 0; K % 32 == 0; A row stride == K (f32);
// B is packed f16 column-major [n][k] with n-extent a multiple of 128 and
// row stride K. Double-buffered LDS; no bounds checks in the hot loop.
// Block: 256 threads = 8 waves; tile 16(M) x 128(N); K stepped by 32.
// ---------------------------------------------------------------------------
#define TM 16
#define TN 128
#define TK 32
#define LDT 40   // LDS row pitch in f16 (80B: keeps 16B alignment, offsets banks)

__global__ __launch_bounds__(256) void k_gemm(
    const float* __restrict__ A, const _Float16* __restrict__ B, float* __restrict__ C,
    int M, int Nl, int K, int ldc, float scale, int flags)
{
    __shared__ _Float16 As[2][TM][LDT];
    __shared__ _Float16 Bs[2][TN][LDT];

    const int m0   = blockIdx.x * TM;
    const int n0   = blockIdx.y * TN;
    const int tid  = threadIdx.x;
    const int lane = tid & 31;
    const int wave = tid >> 5;
    const int half = lane >> 4;
    const int mr   = lane & 15;
    const int nc   = (lane & 15) + wave * 16;

    // staging assignments
    const int ar = tid >> 4;            // 0..15  (A row)
    const int ac = (tid & 15) << 1;     // 0..30  (A col pair)
    const int bn = tid >> 2;            // 0..63  (B row; +64 for second half)
    const int bk = (tid & 3) << 3;      // 0,8,16,24 (B k-chunk)

    const int nk = K >> 5;
    int cur = 0;

    // ---- prologue: stage tile 0 ----
    {
        float2 av = *(const float2*)(A + (size_t)(m0 + ar) * K + ac);
        f16x2 ah; ah.x = (_Float16)av.x; ah.y = (_Float16)av.y;
        *(f16x2*)(&As[0][ar][ac]) = ah;
#if HAVE_TDM
        if (wave == 0) {
            tdm_load_b(B + (size_t)n0 * K, (unsigned)(uintptr_t)&Bs[0][0][0], K);
            __builtin_amdgcn_s_wait_tensorcnt(0);
        }
#else
        *(f16x8*)(&Bs[0][bn][bk])      = *(const f16x8*)(B + (size_t)(n0 + bn) * K + bk);
        *(f16x8*)(&Bs[0][bn + 64][bk]) = *(const f16x8*)(B + (size_t)(n0 + bn + 64) * K + bk);
#endif
    }
    __syncthreads();

    v8f acc = {};
    for (int kt = 0; kt < nk; ++kt) {
        const int nxt = cur ^ 1;
        const bool more = (kt + 1 < nk);

        // issue next-tile loads before computing (overlap with WMMA)
        float2 av;
#if !HAVE_TDM
        f16x8 br0, br1;
#endif
        if (more) {
            const int k1 = (kt + 1) << 5;
            av = *(const float2*)(A + (size_t)(m0 + ar) * K + k1 + ac);
#if HAVE_TDM
            if (wave == 0)
                tdm_load_b(B + (size_t)n0 * K + k1, (unsigned)(uintptr_t)&Bs[nxt][0][0], K);
#else
            br0 = *(const f16x8*)(B + (size_t)(n0 + bn) * K + k1 + bk);
            br1 = *(const f16x8*)(B + (size_t)(n0 + bn + 64) * K + k1 + bk);
#endif
        }

        // compute from current buffers
        f16x8 alo = *(const f16x8*)(&As[cur][mr][8 * half]);
        f16x8 ahi = *(const f16x8*)(&As[cur][mr][16 + 8 * half]);
        f16x8 blo = *(const f16x8*)(&Bs[cur][nc][8 * half]);
        f16x8 bhi = *(const f16x8*)(&Bs[cur][nc][16 + 8 * half]);
        v16h va = __builtin_shufflevector(alo, ahi, 0,1,2,3,4,5,6,7,8,9,10,11,12,13,14,15);
        v16h vb = __builtin_shufflevector(blo, bhi, 0,1,2,3,4,5,6,7,8,9,10,11,12,13,14,15);
        acc = __builtin_amdgcn_wmma_f32_16x16x32_f16(
            false, va, false, vb, (short)0, acc, false, false);

        // commit next tile
        if (more) {
            f16x2 ah; ah.x = (_Float16)av.x; ah.y = (_Float16)av.y;
            *(f16x2*)(&As[nxt][ar][ac]) = ah;
#if HAVE_TDM
            if (wave == 0) __builtin_amdgcn_s_wait_tensorcnt(0);
#else
            *(f16x8*)(&Bs[nxt][bn][bk])      = br0;
            *(f16x8*)(&Bs[nxt][bn + 64][bk]) = br1;
#endif
        }
        __syncthreads();
        cur = nxt;
    }

    const int col = n0 + nc;
    if (col < Nl) {
        union { v8f v; float f[8]; } uc;
        uc.v = acc;
#pragma unroll
        for (int r = 0; r < 8; ++r) {
            int row = m0 + r + 8 * half;
            float v = uc.f[r] * scale;
            size_t o = (size_t)row * ldc + col;
            if (flags & F_ACC)  v += C[o];
            if (flags & F_SILU) v = siluf(v);
            C[o] = v;
        }
    }
}

// ---------------------------------------------------------------------------
// Edge geometry: rvec, rn, bessel*cutoff (+ analytic d/drn), Y1.
// rb stored with K-stride 32 (zero padded) so it can feed the GEMM directly.
// ---------------------------------------------------------------------------
__global__ __launch_bounds__(256) void k_edge_geom(
    const float* __restrict__ pos, const int* __restrict__ snd, const int* __restrict__ rcv,
    float* __restrict__ rb, float* __restrict__ drb, float* __restrict__ Y1,
    float* __restrict__ rvec, float* __restrict__ rno, int E)
{
    int e = blockIdx.x * 256 + threadIdx.x;
    if (e >= E) return;
    int s = snd[e], r = rcv[e];
    float dx = pos[3 * s + 0] - pos[3 * r + 0];
    float dy = pos[3 * s + 1] - pos[3 * r + 1];
    float dz = pos[3 * s + 2] - pos[3 * r + 2];
    float r2 = dx * dx + dy * dy + dz * dz;
    bool  z  = (r2 == 0.0f);
    float rs = sqrtf(z ? 1.0f : r2);
    float rn = z ? 0.0f : rs;

    float x = rn * (1.0f / RMAXf);
    float u = 0.0f, du = 0.0f;
    if (x < 1.0f) {
        float x2 = x * x;
        u  = 1.0f - 6.0f * x2 + 8.0f * x2 * x - 3.0f * x2 * x2;
        du = (-12.0f * x + 24.0f * x2 - 12.0f * x2 * x) * (1.0f / RMAXf);
    }
    float inv = z ? 0.0f : (1.0f / rn);
#pragma unroll
    for (int k = 0; k < 8; ++k) {
        float w = (float)(k + 1) * PIf;
        float b, db;
        if (z) { b = (2.0f / RMAXf) * (w / RMAXf); db = 0.0f; }
        else {
            float a  = w * rn * (1.0f / RMAXf);
            float sn = __sinf(a), cs = __cosf(a);
            b  = (2.0f / RMAXf) * sn * inv;
            db = (2.0f / RMAXf) * (cs * (w / RMAXf) * inv - sn * inv * inv);
        }
        rb[(size_t)e * 32 + k] = b * u;
        drb[e * 8 + k]         = db * u + b * du;
    }
    float q = SQRT3f * (z ? 1.0f : inv);
    Y1[3 * e + 0] = q * dx; Y1[3 * e + 1] = q * dy; Y1[3 * e + 2] = q * dz;
    rvec[3 * e + 0] = dx;   rvec[3 * e + 1] = dy;   rvec[3 * e + 2] = dz;
    rno[e] = rn;
}

// --------------------------- elementwise helpers ---------------------------
__global__ __launch_bounds__(256) void k_silu(const float* __restrict__ x, float* __restrict__ y, int n) {
    int i = blockIdx.x * 256 + threadIdx.x;
    if (i < n) y[i] = siluf(x[i]);
}
__global__ __launch_bounds__(256) void k_silu_bwd(float* __restrict__ g, const float* __restrict__ p, int n) {
    int i = blockIdx.x * 256 + threadIdx.x;
    if (i < n) g[i] *= dsiluf(p[i]);
}

// layer-0: s is one-hot -> s1 = w1_s[species]/sqrt(8)   (N x 8)
__global__ __launch_bounds__(256) void k_s1_l0(
    const int* __restrict__ sp, const float* __restrict__ w1s, float* __restrict__ s1, int N) {
    int i = blockIdx.x * 256 + threadIdx.x;
    if (i >= N * 8) return;
    int n = i >> 3, m = i & 7;
    s1[i] = w1s[sp[n] * 8 + m] * INVSQ8;
}
// layer-0 skip: pre[n,o] += wsk_s[sp,sp,o]/sqrt(8)
__global__ __launch_bounds__(256) void k_skip_l0(
    const int* __restrict__ sp, const float* __restrict__ wsks, float* __restrict__ pre, int N) {
    int i = blockIdx.x * 256 + threadIdx.x;
    if (i >= N * 256) return;
    int n = i >> 8, o = i & 255, s = sp[n];
    pre[i] += wsks[((size_t)s * 8 + s) * 256 + o] * INVSQ8;
}

// --------------------------- edge message passing ---------------------------
// layer-0 forward: 8 channels; agg buffers have row stride 32 (GEMM-ready).
__global__ __launch_bounds__(256) void k_edge_fwd_l0(
    const float* __restrict__ s1, const float* __restrict__ Y1, const float* __restrict__ R16,
    const int* __restrict__ snd, const int* __restrict__ rcv,
    float* __restrict__ aggs, float* __restrict__ aggv, int E)
{
    int i = blockIdx.x * 256 + threadIdx.x;
    if (i >= E * 8) return;
    int e = i >> 3, m = i & 7;
    int s = snd[e], r = rcv[e];
    float val = s1[s * 8 + m];
    float Rs = R16[(size_t)e * 16 + m], Rv = R16[(size_t)e * 16 + 8 + m];
    atomAddF(&aggs[(size_t)r * 32 + m], val * Rs * AINV);
    size_t pst = (size_t)NP * 32;
#pragma unroll
    for (int k = 0; k < 3; ++k)
        atomAddF(&aggv[k * pst + (size_t)r * 32 + m], val * Y1[3 * e + k] * Rv * AINV);
}

// layers 1/2 forward: 256 channels; one block per edge.
__global__ __launch_bounds__(256) void k_edge_fwd(
    const float* __restrict__ s1, const float* __restrict__ v1,
    const float* __restrict__ Y1, const float* __restrict__ R,
    const int* __restrict__ snd, const int* __restrict__ rcv,
    float* __restrict__ aggs, float* __restrict__ aggv, int E)
{
    int e = blockIdx.x;
    int m = threadIdx.x;
    if (e >= E) return;
    int s = snd[e], r = rcv[e];
    float y0 = Y1[3 * e + 0], y1 = Y1[3 * e + 1], y2 = Y1[3 * e + 2];
    const float* Re = R + (size_t)e * 512;
    if (e + 8 < E) __builtin_prefetch((const void*)(Re + 8 * 512), 0, 1);
    float Rs = Re[m], Rv = Re[256 + m];
    size_t vst = (size_t)NP * Hh;
    size_t ast = (size_t)NP * 256;
    if (m < Hh) {
        float val = s1[(size_t)s * Hh + m];
        atomAddF(&aggs[(size_t)r * 256 + m], val * Rs * AINV);
        float t = val * Rv * AINV;
        atomAddF(&aggv[0 * ast + (size_t)r * 256 + m], t * y0);
        atomAddF(&aggv[1 * ast + (size_t)r * 256 + m], t * y1);
        atomAddF(&aggv[2 * ast + (size_t)r * 256 + m], t * y2);
    } else {
        int mm = m - Hh;
        float w0 = v1[0 * vst + (size_t)s * Hh + mm];
        float w1 = v1[1 * vst + (size_t)s * Hh + mm];
        float w2 = v1[2 * vst + (size_t)s * Hh + mm];
        float tps = (w0 * y0 + w1 * y1 + w2 * y2) * INV_SQRT3f;
        atomAddF(&aggs[(size_t)r * 256 + m], tps * Rs * AINV);
        atomAddF(&aggv[0 * ast + (size_t)r * 256 + m], w0 * Rv * AINV);
        atomAddF(&aggv[1 * ast + (size_t)r * 256 + m], w1 * Rv * AINV);
        atomAddF(&aggv[2 * ast + (size_t)r * 256 + m], w2 * Rv * AINV);
    }
}

// --------------------------- per-node skip (GEMV) ---------------------------
__global__ __launch_bounds__(256) void k_skip(
    const float* __restrict__ s_in, const float* __restrict__ v_in,
    const int* __restrict__ sp, const float* __restrict__ wsks, const float* __restrict__ wskv,
    float* __restrict__ pre, float* __restrict__ ov, int N)
{
    __shared__ float sh[Hh];
    int n = blockIdx.x, tid = threadIdx.x;
    int s = sp[n];
    if (tid < Hh) sh[tid] = s_in[(size_t)n * Hh + tid];
    __syncthreads();
    const float* Wk = wsks + (size_t)s * Hh * 256;
    float acc = 0.0f;
    for (int m = 0; m < Hh; ++m) acc += sh[m] * Wk[(size_t)m * 256 + tid];
    pre[(size_t)n * 256 + tid] += acc * INVSQH;
    size_t vst = (size_t)NP * Hh;
    for (int i = 0; i < 3; ++i) {
        __syncthreads();
        if (tid < Hh) sh[tid] = v_in[i * vst + (size_t)n * Hh + tid];
        __syncthreads();
        if (tid < Hh) {
            const float* Wv = wskv + (size_t)s * Hh * Hh;
            float a = 0.0f;
            for (int m = 0; m < Hh; ++m) a += sh[m] * Wv[(size_t)m * Hh + tid];
            ov[i * vst + (size_t)n * Hh + tid] += a * INVSQH;
        }
    }
}

__global__ __launch_bounds__(256) void k_skip_bwd(
    const float* __restrict__ g_pre, const float* __restrict__ g_ov,
    const int* __restrict__ sp, const float* __restrict__ wsks, const float* __restrict__ wskv,
    float* __restrict__ g_sin, float* __restrict__ g_vin, int N)
{
    __shared__ float gh[256];
    int n = blockIdx.x, tid = threadIdx.x;
    int s = sp[n];
    gh[tid] = g_pre[(size_t)n * 256 + tid];
    __syncthreads();
    if (tid < Hh) {
        const float* Wk = wsks + (size_t)s * Hh * 256;
        float a = 0.0f;
        for (int o = 0; o < 256; ++o) a += gh[o] * Wk[(size_t)tid * 256 + o];
        g_sin[(size_t)n * Hh + tid] += a * INVSQH;
    }
    size_t vst = (size_t)NP * Hh;
    for (int i = 0; i < 3; ++i) {
        __syncthreads();
        if (tid < Hh) gh[tid] = g_ov[i * vst + (size_t)n * Hh + tid];
        __syncthreads();
        if (tid < Hh) {
            const float* Wv = wskv + (size_t)s * Hh * Hh;
            float a = 0.0f;
            for (int o = 0; o < Hh; ++o) a += gh[o] * Wv[(size_t)tid * Hh + o];
            g_vin[i * vst + (size_t)n * Hh + tid] += a * INVSQH;
        }
    }
}

// --------------------------- gating ---------------------------
__global__ __launch_bounds__(256) void k_gate(
    const float* __restrict__ pre, const float* __restrict__ ov,
    float* __restrict__ sout, float* __restrict__ vout, int N)
{
    int i = blockIdx.x * 256 + threadIdx.x;
    if (i >= N * Hh) return;
    int n = i / Hh, c = i % Hh;
    sout[i] = siluf(pre[(size_t)n * 256 + c]);
    float g = siluf(pre[(size_t)n * 256 + Hh + c]);
    size_t vst = (size_t)NP * Hh;
#pragma unroll
    for (int k = 0; k < 3; ++k) vout[k * vst + i] = ov[k * vst + i] * g;
}

__global__ __launch_bounds__(256) void k_gate_bwd(
    const float* __restrict__ gs, const float* __restrict__ gv,
    const float* __restrict__ pre, const float* __restrict__ ov,
    float* __restrict__ g_pre, float* __restrict__ g_ov, int N)
{
    int i = blockIdx.x * 256 + threadIdx.x;
    if (i >= N * Hh) return;
    int n = i / Hh, c = i % Hh;
    g_pre[(size_t)n * 256 + c] = gs[i] * dsiluf(pre[(size_t)n * 256 + c]);
    float pg   = pre[(size_t)n * 256 + Hh + c];
    float gate = siluf(pg);
    size_t vst = (size_t)NP * Hh;
    float gg = 0.0f;
#pragma unroll
    for (int k = 0; k < 3; ++k) {
        float gvi = gv[k * vst + i];
        g_ov[k * vst + i] = gvi * gate;
        gg += gvi * ov[k * vst + i];
    }
    g_pre[(size_t)n * 256 + Hh + c] = gg * dsiluf(pg);
}

// --------------------------- readout ---------------------------
__global__ __launch_bounds__(256) void k_energy(
    const float* __restrict__ s3, const float* __restrict__ w_out,
    const float* __restrict__ atomE, const int* __restrict__ sp,
    const int* __restrict__ gidx, float* __restrict__ out, int N)
{
    int n = blockIdx.x * 256 + threadIdx.x;
    if (n >= N) return;
    float acc = 0.0f;
    for (int c = 0; c < Hh; ++c) acc += s3[(size_t)n * Hh + c] * w_out[c];
    float e = acc * INVSQH + atomE[sp[n]];
    atomAddF(&out[gidx[n]], e);
}
__global__ __launch_bounds__(256) void k_init_gs(
    const float* __restrict__ w_out, float* __restrict__ gs, int N) {
    int i = blockIdx.x * 256 + threadIdx.x;
    if (i < N * Hh) gs[i] = w_out[i & (Hh - 1)] * INVSQH;
}

// --------------------------- edge backward ---------------------------
__global__ __launch_bounds__(256) void k_edge_bwd(
    const float* __restrict__ s1, const float* __restrict__ v1,
    const float* __restrict__ Y1, const float* __restrict__ R,
    const int* __restrict__ snd, const int* __restrict__ rcv,
    const float* __restrict__ gAs, const float* __restrict__ gAv,
    float* __restrict__ g_s1, float* __restrict__ g_v1,
    float* __restrict__ gY1, float* __restrict__ gR, int E)
{
    int e = blockIdx.x;
    int m = threadIdx.x;
    if (e >= E) return;
    int s = snd[e], r = rcv[e];
    float y0 = Y1[3 * e + 0], y1 = Y1[3 * e + 1], y2 = Y1[3 * e + 2];
    const float* Re = R + (size_t)e * 512;
    float Rs = Re[m], Rv = Re[256 + m];
    size_t vst = (size_t)NP * Hh;
    size_t ast = (size_t)NP * 256;
    float as  = AINV * gAs[(size_t)r * 256 + m];
    float av0 = AINV * gAv[0 * ast + (size_t)r * 256 + m];
    float av1 = AINV * gAv[1 * ast + (size_t)r * 256 + m];
    float av2 = AINV * gAv[2 * ast + (size_t)r * 256 + m];
    float* gRe = gR + (size_t)e * 512;
    if (m < Hh) {
        float val = s1[(size_t)s * Hh + m];
        float gts = as * Rs;
        gRe[m]       = as * val;
        float gt0 = av0 * Rv, gt1 = av1 * Rv, gt2 = av2 * Rv;
        gRe[256 + m] = (av0 * y0 + av1 * y1 + av2 * y2) * val;
        float gval = gts + gt0 * y0 + gt1 * y1 + gt2 * y2;
        atomAddF(&g_s1[(size_t)s * Hh + m], gval);
        atomAddF(&gY1[3 * e + 0], gt0 * val);
        atomAddF(&gY1[3 * e + 1], gt1 * val);
        atomAddF(&gY1[3 * e + 2], gt2 * val);
    } else {
        int mm = m - Hh;
        float w0 = v1[0 * vst + (size_t)s * Hh + mm];
        float w1 = v1[1 * vst + (size_t)s * Hh + mm];
        float w2 = v1[2 * vst + (size_t)s * Hh + mm];
        float tps = (w0 * y0 + w1 * y1 + w2 * y2) * INV_SQRT3f;
        gRe[m]       = as * tps;
        gRe[256 + m] = av0 * w0 + av1 * w1 + av2 * w2;
        float gts = as * Rs;
        float gt0 = av0 * Rv, gt1 = av1 * Rv, gt2 = av2 * Rv;
        atomAddF(&g_v1[0 * vst + (size_t)s * Hh + mm], gts * y0 * INV_SQRT3f + gt0);
        atomAddF(&g_v1[1 * vst + (size_t)s * Hh + mm], gts * y1 * INV_SQRT3f + gt1);
        atomAddF(&g_v1[2 * vst + (size_t)s * Hh + mm], gts * y2 * INV_SQRT3f + gt2);
        atomAddF(&gY1[3 * e + 0], gts * w0 * INV_SQRT3f);
        atomAddF(&gY1[3 * e + 1], gts * w1 * INV_SQRT3f);
        atomAddF(&gY1[3 * e + 2], gts * w2 * INV_SQRT3f);
    }
}

// layer-0 edge backward: gAs row stride 8, gAv plane stride NP*8;
// gR written with row stride 32 (GEMM-ready; cols 16..31 pre-zeroed).
__global__ __launch_bounds__(256) void k_edge_bwd_l0(
    const float* __restrict__ s1, const float* __restrict__ Y1, const float* __restrict__ R16,
    const int* __restrict__ snd, const int* __restrict__ rcv,
    const float* __restrict__ gAs, const float* __restrict__ gAv,
    float* __restrict__ gY1, float* __restrict__ gR32, int E)
{
    int i = blockIdx.x * 256 + threadIdx.x;
    if (i >= E * 8) return;
    int e = i >> 3, m = i & 7;
    int s = snd[e], r = rcv[e];
    float val = s1[s * 8 + m];
    float Rv = R16[(size_t)e * 16 + 8 + m];
    size_t pst = (size_t)NP * 8;
    float as  = AINV * gAs[(size_t)r * 8 + m];
    float gRv = 0.0f;
#pragma unroll
    for (int k = 0; k < 3; ++k) {
        float av = AINV * gAv[k * pst + (size_t)r * 8 + m];
        gRv += av * val * Y1[3 * e + k];
        atomAddF(&gY1[3 * e + k], av * Rv * val);
    }
    gR32[(size_t)e * 32 + m]     = as * val;
    gR32[(size_t)e * 32 + 8 + m] = gRv;
}

// --------------------------- geometric chain rule ---------------------------
__global__ __launch_bounds__(256) void k_pos_grad(
    const float* __restrict__ grb, const float* __restrict__ drb,
    const float* __restrict__ gY1, const float* __restrict__ rvec, const float* __restrict__ rn,
    const int* __restrict__ snd, const int* __restrict__ rcv, float* __restrict__ gpos, int E)
{
    int e = blockIdx.x * 256 + threadIdx.x;
    if (e >= E) return;
    float rr = rn[e];
    if (rr <= 0.0f) return;
    float grn = 0.0f;
#pragma unroll
    for (int k = 0; k < 8; ++k) grn += grb[e * 8 + k] * drb[e * 8 + k];
    float inv = 1.0f / rr;
    float rx = rvec[3 * e + 0], ry = rvec[3 * e + 1], rz = rvec[3 * e + 2];
    float g0 = gY1[3 * e + 0], g1 = gY1[3 * e + 1], g2 = gY1[3 * e + 2];
    float dot = g0 * rx + g1 * ry + g2 * rz;
    float i3 = inv * inv * inv;
    float gx = grn * rx * inv + SQRT3f * (g0 * inv - dot * rx * i3);
    float gy = grn * ry * inv + SQRT3f * (g1 * inv - dot * ry * i3);
    float gz = grn * rz * inv + SQRT3f * (g2 * inv - dot * rz * i3);
    int s = snd[e], r = rcv[e];
    atomAddF(&gpos[3 * s + 0],  gx); atomAddF(&gpos[3 * s + 1],  gy); atomAddF(&gpos[3 * s + 2],  gz);
    atomAddF(&gpos[3 * r + 0], -gx); atomAddF(&gpos[3 * r + 1], -gy); atomAddF(&gpos[3 * r + 2], -gz);
}

__global__ __launch_bounds__(256) void k_forces(
    const float* __restrict__ gpos, float* __restrict__ out, int n) {
    int i = blockIdx.x * 256 + threadIdx.x;
    if (i < n) out[i] = -gpos[i];
}

// ===========================================================================
// Host side
// ===========================================================================
struct GW { const _Float16* p; int K; };   // packed weight + K stride
struct LayerP {
    const float *w1s, *w1v, *w2s, *w2v, *wr0, *wr1, *wr2, *wr3, *wsks, *wskv;
};
struct LayerG {
    GW w1sN, w1vN, w1sT, w1vT, w2sN, w2vN, w2sT, w2vT,
       wr0N, wr1N, wr2N, wr3N, wr0T, wr1T, wr2T, wr3T;
};

static inline int cdiv(int a, int b) { return (a + b - 1) / b; }
static inline int RU(int a, int b)   { return cdiv(a, b) * b; }

static void gemm(const float* A, GW B, float* C, int Mp, int Nl, int ldc,
                 float scale, int flags, hipStream_t s)
{
    dim3 g(Mp / TM, cdiv(Nl, TN));
    k_gemm<<<g, 256, 0, s>>>(A, B.p, C, Mp, Nl, B.K, ldc, scale, flags);
}

extern "C" void kernel_launch(void* const* d_in, const int* in_sizes, int n_in,
                              void* d_out, int out_size, void* d_ws, size_t ws_size,
                              hipStream_t stream)
{
    (void)in_sizes; (void)n_in; (void)out_size; (void)ws_size;
    const int N = Nn, E = Ee;

    const float* positions = (const float*)d_in[0];
    const float* atomE     = (const float*)d_in[1];
    LayerP L[3];
    L[0] = { (const float*)d_in[2], nullptr,
             (const float*)d_in[3], (const float*)d_in[4],
             (const float*)d_in[5], (const float*)d_in[6], (const float*)d_in[7], (const float*)d_in[8],
             (const float*)d_in[9], nullptr };
    for (int l = 1; l < 3; ++l) {
        int b = 10 * l;
        L[l] = { (const float*)d_in[b + 0], (const float*)d_in[b + 1],
                 (const float*)d_in[b + 2], (const float*)d_in[b + 3],
                 (const float*)d_in[b + 4], (const float*)d_in[b + 5],
                 (const float*)d_in[b + 6], (const float*)d_in[b + 7],
                 (const float*)d_in[b + 8], (const float*)d_in[b + 9] };
    }
    const float* w_out = (const float*)d_in[30];
    const int* species = (const int*)d_in[31];
    const int* snd     = (const int*)d_in[32];
    const int* rcv     = (const int*)d_in[33];
    const int* gidx    = (const int*)d_in[34];
    float* out = (float*)d_out;

    // ---- workspace bump allocator (16B-aligned) ----
    float* base = (float*)d_ws;
    size_t off = 0;
    auto alloc = [&](size_t n) { float* p = base + off; off += (n + 3) & ~(size_t)3; return p; };
    auto allocH = [&](size_t nh) {
        _Float16* p = (_Float16*)(base + off);
        size_t nf = (nh + 1) / 2;
        off += (nf + 3) & ~(size_t)3;
        return p;
    };
    const size_t NH = (size_t)NP * Hh, N256 = (size_t)NP * 256, E64 = (size_t)E * 64;

    // ---- pack all GEMM weights to padded f16 (normal: col-major; trans: row-major)
    auto packN = [&](const float* W, int Kf, int Nf) -> GW {
        int Ro = RU(Nf, 128), Co = RU(Kf, 32);
        _Float16* o = allocH((size_t)Ro * Co);
        k_pack<<<cdiv(Ro * Co, 256), 256, 0, stream>>>(W, o, Kf, Nf, Ro, Co, 1);
        return { o, Co };
    };
    auto packT = [&](const float* W, int Kf, int Nf) -> GW {
        int Ro = RU(Kf, 128), Co = RU(Nf, 32);
        _Float16* o = allocH((size_t)Ro * Co);
        k_pack<<<cdiv(Ro * Co, 256), 256, 0, stream>>>(W, o, Kf, Nf, Ro, Co, 0);
        return { o, Co };
    };

    LayerG G[3];
    // layer 0 (w1_s consumed by gather kernel, not GEMM)
    G[0].w2sN = packN(L[0].w2s, 8, 256);   G[0].w2sT = packT(L[0].w2s, 8, 256);
    G[0].w2vN = packN(L[0].w2v, 8, 128);   G[0].w2vT = packT(L[0].w2v, 8, 128);
    G[0].wr0N = packN(L[0].wr0, 8, 64);    G[0].wr0T = packT(L[0].wr0, 8, 64);
    G[0].wr1N = packN(L[0].wr1, 64, 64);   G[0].wr1T = packT(L[0].wr1, 64, 64);
    G[0].wr2N = packN(L[0].wr2, 64, 64);   G[0].wr2T = packT(L[0].wr2, 64, 64);
    G[0].wr3N = packN(L[0].wr3, 64, 16);   G[0].wr3T = packT(L[0].wr3, 64, 16);
    for (int l = 1; l < 3; ++l) {
        G[l].w1sN = packN(L[l].w1s, 128, 128); G[l].w1sT = packT(L[l].w1s, 128, 128);
        G[l].w1vN = packN(L[l].w1v, 128, 128); G[l].w1vT = packT(L[l].w1v, 128, 128);
        G[l].w2sN = packN(L[l].w2s, 256, 256); G[l].w2sT = packT(L[l].w2s, 256, 256);
        G[l].w2vN = packN(L[l].w2v, 256, 128); G[l].w2vT = packT(L[l].w2v, 256, 128);
        G[l].wr0N = packN(L[l].wr0, 8, 64);    G[l].wr0T = packT(L[l].wr0, 8, 64);
        G[l].wr1N = packN(L[l].wr1, 64, 64);   G[l].wr1T = packT(L[l].wr1, 64, 64);
        G[l].wr2N = packN(L[l].wr2, 64, 64);   G[l].wr2T = packT(L[l].wr2, 64, 64);
        G[l].wr3N = packN(L[l].wr3, 64, 512);  G[l].wr3T = packT(L[l].wr3, 64, 512);
    }

    // ---- activations / gradients ----
    float* rb   = alloc((size_t)E * 32);        // K-padded
    float* drb  = alloc((size_t)E * 8);
    float* Y1   = alloc((size_t)E * 3);
    float* rvec = alloc((size_t)E * 3);
    float* rn   = alloc((size_t)E);
    float* gY1  = alloc((size_t)E * 3);
    float* grb  = alloc((size_t)E * 8);
    float* Rbuf = alloc((size_t)E * 512);
    float* gR   = alloc((size_t)E * 512);
    float* p1 = alloc(E64); float* p2 = alloc(E64); float* p3 = alloc(E64);
    float* h1 = alloc(E64); float* h2 = alloc(E64); float* h3 = alloc(E64);
    float* gb1 = alloc(E64); float* gb2 = alloc(E64);
    float* s1l0  = alloc((size_t)N * 8);
    float* aggs8 = alloc((size_t)NP * 32);
    float* aggv8 = alloc((size_t)NP * 32 * 3);
    float* preB[3], *ovB[3], *soutB[3], *voutB[3];
    for (int l = 0; l < 3; ++l) { preB[l] = alloc(N256); ovB[l] = alloc(3 * NH);
                                  soutB[l] = alloc(NH);  voutB[l] = alloc(3 * NH); }
    float* s1s[2]; float* v1s[2];
    for (int l = 0; l < 2; ++l) { s1s[l] = alloc(NH); v1s[l] = alloc(3 * NH); }
    float* aggs  = alloc(N256);
    float* aggv  = alloc(3 * N256);
    float* gsA   = alloc(NH);      float* gvA   = alloc(3 * NH);
    float* gsB   = alloc(NH);      float* gvB   = alloc(3 * NH);
    float* g_pre = alloc(N256);    float* g_ov  = alloc(3 * NH);
    float* g_ags = alloc(N256);    float* g_agv = alloc(3 * N256);
    float* g_s1  = alloc(NH);      float* g_v1  = alloc(3 * NH);
    float* gpos  = alloc((size_t)N * 3);

    // ======================= FORWARD =======================
    hipMemsetAsync(rb, 0, (size_t)E * 32 * 4, stream);   // zero K-pad cols
    k_edge_geom<<<cdiv(E, 256), 256, 0, stream>>>(positions, snd, rcv, rb, drb, Y1, rvec, rn, E);

    // ---- layer 0 (one-hot input, ms=8, no v) ----
    k_s1_l0<<<cdiv(N * 8, 256), 256, 0, stream>>>(species, L[0].w1s, s1l0, N);
    gemm(rb, G[0].wr0N, h1, E, 64, 64, 1.0f, F_SILU, stream);
    gemm(h1, G[0].wr1N, h2, E, 64, 64, 1.0f, F_SILU, stream);
    gemm(h2, G[0].wr2N, h3, E, 64, 64, 1.0f, F_SILU, stream);
    gemm(h3, G[0].wr3N, Rbuf, E, 16, 16, 1.0f, 0, stream);   // R16: E x 16
    hipMemsetAsync(aggs8, 0, (size_t)NP * 32 * 4, stream);
    hipMemsetAsync(aggv8, 0, (size_t)NP * 32 * 3 * 4, stream);
    k_edge_fwd_l0<<<cdiv(E * 8, 256), 256, 0, stream>>>(s1l0, Y1, Rbuf, snd, rcv, aggs8, aggv8, E);
    gemm(aggs8, G[0].w2sN, preB[0], NP, 256, 256, INVSQ8, 0, stream);
    for (int i = 0; i < 3; ++i)
        gemm(aggv8 + (size_t)i * NP * 32, G[0].w2vN, ovB[0] + (size_t)i * NH, NP, 128, 128, INVSQ8, 0, stream);
    k_skip_l0<<<cdiv(N * 256, 256), 256, 0, stream>>>(species, L[0].wsks, preB[0], N);
    k_gate<<<cdiv(N * Hh, 256), 256, 0, stream>>>(preB[0], ovB[0], soutB[0], voutB[0], N);

    // ---- layers 1, 2 ----
    const float* sin_p = soutB[0];
    const float* vin_p = voutB[0];
    for (int l = 0; l < 2; ++l) {
        const LayerG& P = G[l + 1];
        gemm(sin_p, P.w1sN, s1s[l], NP, 128, 128, INVSQH, 0, stream);
        for (int i = 0; i < 3; ++i)
            gemm(vin_p + (size_t)i * NH, P.w1vN, v1s[l] + (size_t)i * NH, NP, 128, 128, INVSQH, 0, stream);
        gemm(rb, P.wr0N, h1, E, 64, 64, 1.0f, F_SILU, stream);
        gemm(h1, P.wr1N, h2, E, 64, 64, 1.0f, F_SILU, stream);
        gemm(h2, P.wr2N, h3, E, 64, 64, 1.0f, F_SILU, stream);
        gemm(h3, P.wr3N, Rbuf, E, 512, 512, 1.0f, 0, stream);
        hipMemsetAsync(aggs, 0, N256 * 4, stream);
        hipMemsetAsync(aggv, 0, 3 * N256 * 4, stream);
        k_edge_fwd<<<E, 256, 0, stream>>>(s1s[l], v1s[l], Y1, Rbuf, snd, rcv, aggs, aggv, E);
        gemm(aggs, P.w2sN, preB[l + 1], NP, 256, 256, W2SC, 0, stream);
        for (int i = 0; i < 3; ++i)
            gemm(aggv + (size_t)i * N256, P.w2vN, ovB[l + 1] + (size_t)i * NH, NP, 128, 128, W2SC, 0, stream);
        k_skip<<<N, 256, 0, stream>>>(sin_p, vin_p, species, L[l + 1].wsks, L[l + 1].wskv,
                                      preB[l + 1], ovB[l + 1], N);
        k_gate<<<cdiv(N * Hh, 256), 256, 0, stream>>>(preB[l + 1], ovB[l + 1], soutB[l + 1], voutB[l + 1], N);
        sin_p = soutB[l + 1];
        vin_p = voutB[l + 1];
    }

    // ---- readout ----
    hipMemsetAsync(out, 0, 4 * sizeof(float), stream);
    k_energy<<<cdiv(N, 256), 256, 0, stream>>>(soutB[2], w_out, atomE, species, gidx, out, N);

    // ======================= BACKWARD =======================
    float *gs_c = gsA, *gv_c = gvA, *gs_n = gsB, *gv_n = gvB;
    k_init_gs<<<cdiv(N * Hh, 256), 256, 0, stream>>>(w_out, gs_c, N);
    hipMemsetAsync(gv_c, 0, 3 * NH * 4, stream);
    hipMemsetAsync(gpos, 0, (size_t)N * 3 * 4, stream);
    hipMemsetAsync(gY1, 0, (size_t)E * 3 * 4, stream);
    hipMemsetAsync(grb, 0, (size_t)E * 8 * 4, stream);

    for (int l = 1; l >= 0; --l) {
        const LayerG& P = G[l + 1];
        k_gate_bwd<<<cdiv(N * Hh, 256), 256, 0, stream>>>(gs_c, gv_c, preB[l + 1], ovB[l + 1], g_pre, g_ov, N);
        hipMemsetAsync(gs_n, 0, NH * 4, stream);
        hipMemsetAsync(gv_n, 0, 3 * NH * 4, stream);
        k_skip_bwd<<<N, 256, 0, stream>>>(g_pre, g_ov, species, L[l + 1].wsks, L[l + 1].wskv, gs_n, gv_n, N);
        gemm(g_pre, P.w2sT, g_ags, NP, 256, 256, W2SC, 0, stream);
        for (int i = 0; i < 3; ++i)
            gemm(g_ov + (size_t)i * NH, P.w2vT, g_agv + (size_t)i * N256, NP, 256, 256, W2SC, 0, stream);
        // recompute radial with pre-activations
        gemm(rb, P.wr0N, p1, E, 64, 64, 1.0f, 0, stream);
        k_silu<<<cdiv((int)E64, 256), 256, 0, stream>>>(p1, h1, (int)E64);
        gemm(h1, P.wr1N, p2, E, 64, 64, 1.0f, 0, stream);
        k_silu<<<cdiv((int)E64, 256), 256, 0, stream>>>(p2, h2, (int)E64);
        gemm(h2, P.wr2N, p3, E, 64, 64, 1.0f, 0, stream);
        k_silu<<<cdiv((int)E64, 256), 256, 0, stream>>>(p3, h3, (int)E64);
        gemm(h3, P.wr3N, Rbuf, E, 512, 512, 1.0f, 0, stream);
        hipMemsetAsync(g_s1, 0, NH * 4, stream);
        hipMemsetAsync(g_v1, 0, 3 * NH * 4, stream);
        k_edge_bwd<<<E, 256, 0, stream>>>(s1s[l], v1s[l], Y1, Rbuf, snd, rcv,
                                          g_ags, g_agv, g_s1, g_v1, gY1, gR, E);
        // radial MLP backward
        gemm(gR, P.wr3T, gb1, E, 64, 64, 1.0f, 0, stream);
        k_silu_bwd<<<cdiv((int)E64, 256), 256, 0, stream>>>(gb1, p3, (int)E64);
        gemm(gb1, P.wr2T, gb2, E, 64, 64, 1.0f, 0, stream);
        k_silu_bwd<<<cdiv((int)E64, 256), 256, 0, stream>>>(gb2, p2, (int)E64);
        gemm(gb2, P.wr1T, gb1, E, 64, 64, 1.0f, 0, stream);
        k_silu_bwd<<<cdiv((int)E64, 256), 256, 0, stream>>>(gb1, p1, (int)E64);
        gemm(gb1, P.wr0T, grb, E, 8, 8, 1.0f, F_ACC, stream);
        // input grads via w1
        gemm(g_s1, P.w1sT, gs_n, NP, 128, 128, INVSQH, F_ACC, stream);
        for (int i = 0; i < 3; ++i)
            gemm(g_v1 + (size_t)i * NH, P.w1vT, gv_n + (size_t)i * NH, NP, 128, 128, INVSQH, F_ACC, stream);
        { float* t = gs_c; gs_c = gs_n; gs_n = t; }
        { float* t = gv_c; gv_c = gv_n; gv_n = t; }
    }

    // ---- layer 0 backward (only R and Y1 paths reach positions) ----
    k_gate_bwd<<<cdiv(N * Hh, 256), 256, 0, stream>>>(gs_c, gv_c, preB[0], ovB[0], g_pre, g_ov, N);
    gemm(g_pre, G[0].w2sT, g_ags, NP, 8, 8, INVSQ8, 0, stream);
    for (int i = 0; i < 3; ++i)
        gemm(g_ov + (size_t)i * NH, G[0].w2vT, g_agv + (size_t)i * NP * 8, NP, 8, 8, INVSQ8, 0, stream);
    gemm(rb, G[0].wr0N, p1, E, 64, 64, 1.0f, 0, stream);
    k_silu<<<cdiv((int)E64, 256), 256, 0, stream>>>(p1, h1, (int)E64);
    gemm(h1, G[0].wr1N, p2, E, 64, 64, 1.0f, 0, stream);
    k_silu<<<cdiv((int)E64, 256), 256, 0, stream>>>(p2, h2, (int)E64);
    gemm(h2, G[0].wr2N, p3, E, 64, 64, 1.0f, 0, stream);
    k_silu<<<cdiv((int)E64, 256), 256, 0, stream>>>(p3, h3, (int)E64);
    gemm(h3, G[0].wr3N, Rbuf, E, 16, 16, 1.0f, 0, stream);
    hipMemsetAsync(gR, 0, (size_t)E * 32 * 4, stream);   // zero pad cols 16..31
    k_edge_bwd_l0<<<cdiv(E * 8, 256), 256, 0, stream>>>(s1l0, Y1, Rbuf, snd, rcv, g_ags, g_agv, gY1, gR, E);
    gemm(gR, G[0].wr3T, gb1, E, 64, 64, 1.0f, 0, stream);
    k_silu_bwd<<<cdiv((int)E64, 256), 256, 0, stream>>>(gb1, p3, (int)E64);
    gemm(gb1, G[0].wr2T, gb2, E, 64, 64, 1.0f, 0, stream);
    k_silu_bwd<<<cdiv((int)E64, 256), 256, 0, stream>>>(gb2, p2, (int)E64);
    gemm(gb2, G[0].wr1T, gb1, E, 64, 64, 1.0f, 0, stream);
    k_silu_bwd<<<cdiv((int)E64, 256), 256, 0, stream>>>(gb1, p1, (int)E64);
    gemm(gb1, G[0].wr0T, grb, E, 8, 8, 1.0f, F_ACC, stream);

    // ---- geometry -> forces ----
    k_pos_grad<<<cdiv(E, 256), 256, 0, stream>>>(grb, drb, gY1, rvec, rn, snd, rcv, gpos, E);
    k_forces<<<cdiv(N * 3, 256), 256, 0, stream>>>(gpos, out + 4, N * 3);
}
